// MENTOR_79774722556360
// MI455X (gfx1250) — matrix operand order repaired and checked by reading, not compiled, not run on hardware
//
#include <hip/hip_runtime.h>
#include <hip/hip_fp16.h>
#include <stdint.h>

#define U_CNT 50000
#define I_CNT 50000
#define N_CNT 100000
#define NE_CNT 1000000

typedef __attribute__((ext_vector_type(16))) _Float16 v16h;
typedef __attribute__((ext_vector_type(8)))  _Float16 v8h;
typedef __attribute__((ext_vector_type(8)))  float    v8f;

__device__ __forceinline__ void atomAddF(float* p, float v) {
  __hip_atomic_fetch_add(p, v, __ATOMIC_RELAXED, __HIP_MEMORY_SCOPE_AGENT);
}

// ---------------- utility kernels ----------------
__global__ void k_zero(float* p, long long n) {
  long long i = (long long)blockIdx.x * blockDim.x + threadIdx.x;
  if (i < n) p[i] = 0.f;
}

__global__ void k_copy(const float* __restrict__ s, float* __restrict__ d, long long n) {
  long long i = (long long)blockIdx.x * blockDim.x + threadIdx.x;
  if (i < n) d[i] = s[i];
}

__global__ void k_deg(const int* __restrict__ row, float* __restrict__ deg, int ne) {
  int e = blockIdx.x * blockDim.x + threadIdx.x;
  if (e < ne) atomAddF(&deg[row[e]], 1.0f);
}

__global__ void k_dinv(float* deg, int n) {
  int i = blockIdx.x * blockDim.x + threadIdx.x;
  if (i < n) deg[i] = (deg[i] > 0.f) ? rsqrtf(deg[i]) : 0.f;
}

__global__ void k_norm(const int* __restrict__ row, const int* __restrict__ col,
                       const float* __restrict__ dinv, float* __restrict__ nrm, int ne) {
  int e = blockIdx.x * blockDim.x + threadIdx.x;
  if (e < ne) nrm[e] = dinv[row[e]] * dinv[col[e]];
}

__global__ void k_f2h(const float* __restrict__ s, __half* __restrict__ d, long long n) {
  long long i = (long long)blockIdx.x * blockDim.x + threadIdx.x;
  if (i < n) d[i] = __float2half(s[i]);
}

// Wt[n*K + k] = W[k*Ncol + n]  (f32 -> f16 transpose)
__global__ void k_wtrans(const float* __restrict__ W, __half* __restrict__ Wt, int K, int Ncol) {
  int i = blockIdx.x * blockDim.x + threadIdx.x;
  int total = K * Ncol;
  if (i >= total) return;
  int n = i / K, k = i % K;
  Wt[i] = __float2half(W[k * Ncol + n]);
}

// ---------------- WMMA GEMM kernels ----------------
// A: [M,K] f16 row-major. Bt: [Ncol,K] f16 row-major (pre-transposed weights).
// One wave computes one 16x16 tile of C.  EXEC is uniform per wave (tile guard).

__device__ __forceinline__ v8f wmma_tile(const _Float16* __restrict__ A,
                                         const _Float16* __restrict__ Bt,
                                         int mrow, int ncol, int K, int lane) {
  int kbA = (lane >> 4) << 4;   // 0 or 16
  int kbB = (lane >> 4) << 4;   // 0 or 16
  v8f c = {};
  for (int k0 = 0; k0 < K; k0 += 32) {
    v8h alo = *(const v8h*)(A + (size_t)mrow * K + k0 + kbA);
    v8h ahi = *(const v8h*)(A + (size_t)mrow * K + k0 + kbA + 8);
    v16h a;
#pragma unroll
    for (int j = 0; j < 8; ++j) { a[j] = alo[j]; a[j + 8] = ahi[j]; }
    v16h b = *(const v16h*)(Bt + (size_t)ncol * K + k0 + kbB);
    c = __builtin_amdgcn_wmma_f32_16x16x32_f16(false, a, false, b, (short)0, c, false, false);
  }
  return c;
}

// C = leaky_relu(A@W + b) stored as f16 [M, Ncol]
__global__ void k_gemm_relu_h16(const __half* __restrict__ Ap, const __half* __restrict__ Btp,
                                const float* __restrict__ bias, __half* __restrict__ Outp,
                                int K, int Ncol, int nTiles) {
  int gtid = blockIdx.x * blockDim.x + threadIdx.x;
  int wave = gtid >> 5, lane = gtid & 31;
  if (wave >= nTiles) return;
  int ntn = Ncol >> 4;
  int mt = wave / ntn, nt = wave % ntn;
  int mrow = (mt << 4) + (lane & 15);
  int ncol = (nt << 4) + (lane & 15);
  v8f c = wmma_tile((const _Float16*)Ap, (const _Float16*)Btp, mrow, ncol, K, lane);
  float bs = bias[ncol];
  int mbase = (mt << 4) + ((lane >> 4) << 3);
  _Float16* Out = (_Float16*)Outp;
#pragma unroll
  for (int r = 0; r < 8; ++r) {
    float v = c[r] + bs;
    v = v > 0.f ? v : 0.01f * v;
    Out[(size_t)(mbase + r) * Ncol + ncol] = (_Float16)v;
  }
}

// C = A@W + b stored as f32 [M, Ncol]
__global__ void k_gemm_f32(const __half* __restrict__ Ap, const __half* __restrict__ Btp,
                           const float* __restrict__ bias, float* __restrict__ Out,
                           int K, int Ncol, int nTiles) {
  int gtid = blockIdx.x * blockDim.x + threadIdx.x;
  int wave = gtid >> 5, lane = gtid & 31;
  if (wave >= nTiles) return;
  int ntn = Ncol >> 4;
  int mt = wave / ntn, nt = wave % ntn;
  int mrow = (mt << 4) + (lane & 15);
  int ncol = (nt << 4) + (lane & 15);
  v8f c = wmma_tile((const _Float16*)Ap, (const _Float16*)Btp, mrow, ncol, K, lane);
  float bs = bias[ncol];
  int mbase = (mt << 4) + ((lane >> 4) << 3);
#pragma unroll
  for (int r = 0; r < 8; ++r)
    Out[(size_t)(mbase + r) * Ncol + ncol] = c[r] + bs;
}

// ---------------- GCN kernels ----------------
// wave per row of 64, 2 floats per lane
__global__ void k_l2norm(float* __restrict__ x, int rows) {
  int gtid = blockIdx.x * blockDim.x + threadIdx.x;
  int r = gtid >> 5, lane = gtid & 31;
  if (r >= rows) return;
  float2 v = *(float2*)(x + (size_t)r * 64 + lane * 2);
  float s = v.x * v.x + v.y * v.y;
#pragma unroll
  for (int m = 16; m; m >>= 1) s += __shfl_xor(s, m, 32);
  float inv = 1.f / fmaxf(sqrtf(s), 1e-12f);
  v.x *= inv; v.y *= inv;
  *(float2*)(x + (size_t)r * 64 + lane * 2) = v;
}

// wave per edge: h[col] += norm * x[row]
__global__ void k_scatter(const int* __restrict__ row, const int* __restrict__ col,
                          const float* __restrict__ nrm, const float* __restrict__ xin,
                          float* __restrict__ hout, int ne) {
  long long gtid = (long long)blockIdx.x * blockDim.x + threadIdx.x;
  int e = (int)(gtid >> 5), lane = (int)(gtid & 31);
  if (e >= ne) return;
  int r = row[e], c = col[e];
  float w = nrm[e];
  float2 v = *(const float2*)(xin + (size_t)r * 64 + lane * 2);
  float* dst = hout + (size_t)c * 64 + lane * 2;
  atomAddF(dst, w * v.x);
  atomAddF(dst + 1, w * v.y);
}

__device__ __forceinline__ float uhash(uint32_t seed, uint32_t a, uint32_t b) {
  uint32_t x = seed * 0x9E3779B1u ^ a * 0x85EBCA77u ^ b * 0xC2B2AE3Du;
  x ^= x >> 16; x *= 0x7FEB352Du; x ^= x >> 15; x *= 0x846CA68Bu; x ^= x >> 16;
  return (float)(x >> 8) * (1.0f / 16777216.0f);
}

// h += sign(h) * 0.1 * l2norm(uniform_row)
__global__ void k_noise(float* __restrict__ h, int rows, uint32_t seed) {
  int gtid = blockIdx.x * blockDim.x + threadIdx.x;
  int r = gtid >> 5, lane = gtid & 31;
  if (r >= rows) return;
  float u0 = uhash(seed, (uint32_t)r, (uint32_t)(2 * lane));
  float u1 = uhash(seed, (uint32_t)r, (uint32_t)(2 * lane + 1));
  float s = u0 * u0 + u1 * u1;
#pragma unroll
  for (int m = 16; m; m >>= 1) s += __shfl_xor(s, m, 32);
  float inv = 1.f / fmaxf(sqrtf(s), 1e-12f);
  float2 hv = *(float2*)(h + (size_t)r * 64 + lane * 2);
  float s0 = hv.x > 0.f ? 1.f : (hv.x < 0.f ? -1.f : 0.f);
  float s1 = hv.y > 0.f ? 1.f : (hv.y < 0.f ? -1.f : 0.f);
  hv.x += s0 * 0.1f * u0 * inv;
  hv.y += s1 * 0.1f * u1 * inv;
  *(float2*)(h + (size_t)r * 64 + lane * 2) = hv;
}

// rep = x+h+h1; write into up to 3 output slices + optional weighted user slice
__global__ void k_writeout(const float* __restrict__ x, const float* __restrict__ h,
                           const float* __restrict__ h1, float* __restrict__ O,
                           int b0, int c0, int b1, int c1, int b2, int c2,
                           const float* __restrict__ wu, int wsel, int wcol) {
  long long i = (long long)blockIdx.x * blockDim.x + threadIdx.x;
  if (i >= (long long)N_CNT * 64) return;
  int n = (int)(i >> 6), d = (int)(i & 63);
  float v = x[i] + h[i] + h1[i];
  if (b0 >= 0) O[(size_t)(b0 + n) * 128 + c0 + d] = v;
  if (b1 >= 0) O[(size_t)(b1 + n) * 128 + c1 + d] = v;
  if (b2 >= 0) O[(size_t)(b2 + n) * 128 + c2 + d] = v;
  if (wu != nullptr && n < U_CNT)
    O[(size_t)(6 * N_CNT + n) * 128 + wcol + d] = v * wu[(size_t)n * 2 + wsel];
}

// ---------------- orchestration ----------------
extern "C" void kernel_launch(void* const* d_in, const int* in_sizes, int n_in,
                              void* d_out, int out_size, void* d_ws, size_t ws_size,
                              hipStream_t stream) {
  (void)in_sizes; (void)n_in; (void)out_size; (void)ws_size;
  const int* edge = (const int*)d_in[0];
  const int* row = edge;
  const int* col = edge + NE_CNT;
  const float* v_feat  = (const float*)d_in[1];
  const float* t_feat  = (const float*)d_in[2];
  const float* id_feat = (const float*)d_in[3];
  const float* pref_v  = (const float*)d_in[4];
  const float* w1_v    = (const float*)d_in[5];
  const float* b1_v    = (const float*)d_in[6];
  const float* w2_v    = (const float*)d_in[7];
  const float* b2_v    = (const float*)d_in[8];
  const float* pref_t  = (const float*)d_in[9];
  const float* w1_t    = (const float*)d_in[10];
  const float* b1_t    = (const float*)d_in[11];
  const float* w2_t    = (const float*)d_in[12];
  const float* b2_t    = (const float*)d_in[13];
  const float* pref_id = (const float*)d_in[14];
  const float* w1_id   = (const float*)d_in[15];
  const float* b1_id   = (const float*)d_in[16];
  const float* w2_id   = (const float*)d_in[17];
  const float* b2_id   = (const float*)d_in[18];
  const float* weight_u = (const float*)d_in[19];
  float* O = (float*)d_out;

  char* ws = (char*)d_ws;
  size_t off = 0;
  auto take = [&](size_t bytes) -> char* {
    char* p = ws + off;
    off = (off + bytes + 255) & ~(size_t)255;
    return p;
  };
  float*  dinv   = (float*)take((size_t)N_CNT * 4);
  float*  nrm    = (float*)take((size_t)NE_CNT * 4);
  float*  x      = (float*)take((size_t)N_CNT * 64 * 4);
  float*  h      = (float*)take((size_t)N_CNT * 64 * 4);
  float*  h1     = (float*)take((size_t)N_CNT * 64 * 4);
  __half* feat16 = (__half*)take((size_t)I_CNT * 128 * 2);
  __half* H16    = (__half*)take((size_t)I_CNT * 256 * 2);
  __half* w1t    = (__half*)take((size_t)256 * 128 * 2);
  __half* w2t    = (__half*)take((size_t)64 * 256 * 2);

  const int B = 256;
  auto blocks = [](long long n) { return (int)((n + 255) / 256); };

  // degree -> dinv -> per-edge norm (shared by all 14 propagates)
  k_zero<<<blocks(N_CNT), B, 0, stream>>>(dinv, N_CNT);
  k_deg<<<blocks(NE_CNT), B, 0, stream>>>(row, dinv, NE_CNT);
  k_dinv<<<blocks(N_CNT), B, 0, stream>>>(dinv, N_CNT);
  k_norm<<<blocks(NE_CNT), B, 0, stream>>>(row, col, dinv, nrm, NE_CNT);

  auto run_branch = [&](const float* pref, const float* feat, int Din,
                        const float* w1, const float* b1, const float* w2, const float* b2,
                        int seed0, int seed1,
                        int bb0, int cc0, int bb1, int cc1, int bb2, int cc2,
                        const float* wu, int wsel, int wcol) {
    // MLP: feat -> H16 (leaky relu) -> temp_feat rows of x
    k_f2h<<<blocks((long long)I_CNT * Din), B, 0, stream>>>(feat, feat16, (long long)I_CNT * Din);
    k_wtrans<<<blocks((long long)Din * 256), B, 0, stream>>>(w1, w1t, Din, 256);
    k_wtrans<<<blocks((long long)256 * 64), B, 0, stream>>>(w2, w2t, 256, 64);
    int t1 = (I_CNT / 16) * (256 / 16);
    k_gemm_relu_h16<<<(t1 + 7) / 8, 256, 0, stream>>>(feat16, w1t, b1, H16, Din, 256, t1);
    int t2 = (I_CNT / 16) * (64 / 16);
    k_gemm_f32<<<(t2 + 7) / 8, 256, 0, stream>>>(H16, w2t, b2, x + (size_t)U_CNT * 64, 256, 64, t2);
    // x = l2norm(concat(pref, temp_feat))
    k_copy<<<blocks((long long)U_CNT * 64), B, 0, stream>>>(pref, x, (long long)U_CNT * 64);
    k_l2norm<<<blocks((long long)N_CNT * 32), B, 0, stream>>>(x, N_CNT);
    // h = propagate(x) (+ noise)
    k_zero<<<blocks((long long)N_CNT * 64), B, 0, stream>>>(h, (long long)N_CNT * 64);
    k_scatter<<<blocks((long long)NE_CNT * 32), B, 0, stream>>>(row, col, nrm, x, h, NE_CNT);
    if (seed0 >= 0) k_noise<<<blocks((long long)N_CNT * 32), B, 0, stream>>>(h, N_CNT, (uint32_t)seed0);
    // h1 = propagate(h) (+ noise)
    k_zero<<<blocks((long long)N_CNT * 64), B, 0, stream>>>(h1, (long long)N_CNT * 64);
    k_scatter<<<blocks((long long)NE_CNT * 32), B, 0, stream>>>(row, col, nrm, h, h1, NE_CNT);
    if (seed1 >= 0) k_noise<<<blocks((long long)N_CNT * 32), B, 0, stream>>>(h1, N_CNT, (uint32_t)seed1);
    // rep = x + h + h1 -> output slices
    k_writeout<<<blocks((long long)N_CNT * 64), B, 0, stream>>>(
        x, h, h1, O, bb0, cc0, bb1, cc1, bb2, cc2, wu, wsel, wcol);
  };

  const int N = N_CNT;
  const size_t pS = (size_t)U_CNT * 64;     // pref slice stride
  const size_t w1vS = (size_t)128 * 256;    // w1_v slice stride
  const size_t w1tS = (size_t)64 * 256;     // w1_t slice stride
  const size_t w2S  = (size_t)256 * 64;     // w2 slice stride

  // v_rep: rep[:, :64], v_repr both halves, user left half (w0)
  run_branch(pref_v, v_feat, 128, w1_v, b1_v, w2_v, b2_v, -1, -1,
             0, 0, 2 * N, 0, 2 * N, 64, weight_u, 0, 0);
  // t_rep: rep[:, 64:], t_repr both halves, user right half (w1)
  run_branch(pref_t, t_feat, 64, w1_t, b1_t, w2_t, b2_t, -1, -1,
             0, 64, 3 * N, 0, 3 * N, 64, weight_u, 1, 64);
  // id_rep: guide both halves
  run_branch(pref_id, id_feat, 64, w1_id, b1_id, w2_id, b2_id, -1, -1,
             1 * N, 0, 1 * N, 64, -1, 0, nullptr, 0, 0);
  // v_n1 / t_n1 -> rep_n1
  run_branch(pref_v + pS, v_feat, 128, w1_v + w1vS, b1_v + 256, w2_v + w2S, b2_v + 64,
             100, 101, 4 * N, 0, -1, 0, -1, 0, nullptr, 0, 0);
  run_branch(pref_t + pS, t_feat, 64, w1_t + w1tS, b1_t + 256, w2_t + w2S, b2_t + 64,
             200, 201, 4 * N, 64, -1, 0, -1, 0, nullptr, 0, 0);
  // v_n2 / t_n2 -> rep_n2
  run_branch(pref_v + 2 * pS, v_feat, 128, w1_v + 2 * w1vS, b1_v + 512, w2_v + 2 * w2S, b2_v + 128,
             300, 301, 5 * N, 0, -1, 0, -1, 0, nullptr, 0, 0);
  run_branch(pref_t + 2 * pS, t_feat, 64, w1_t + 2 * w1tS, b1_t + 512, w2_t + 2 * w2S, b2_t + 128,
             400, 401, 5 * N, 64, -1, 0, -1, 0, nullptr, 0, 0);
}
